// TransformerBlock_58188216926707
// MI455X (gfx1250) — matrix operand order repaired and checked by reading, not compiled, not run on hardware
//
#include <hip/hip_runtime.h>
#include <hip/hip_bf16.h>

// ---------------------------------------------------------------------------
// Transformer block (pre-LN, causal MHA, FFN) for B=2,S=2048,D=1024,H=16.
// GEMMs: v_wmma_f32_16x16x32_bf16, LDS double-buffered tiles filled with
// global_load_async_to_lds_b128 (ASYNCcnt pipelining). f32 accumulate.
// ---------------------------------------------------------------------------

#define BATCH   2
#define SEQ     2048
#define DMODEL  1024
#define NHEAD   16
#define HDIM    64
#define DFFN    4096
#define MROWS   (BATCH * SEQ)     // 4096
#define SPAD    (SEQ + 32)        // padded key stride for Vt (2080)

#define TILE_M  128               // workgroup C-tile rows
#define TILE_N  64                // workgroup C-tile cols
#define TILE_K  64                // K staged per LDS buffer
#define LDS_PITCH 72              // 64 + 8 pad elems (36 dwords -> no bank hot spots)

typedef __bf16 v16bf __attribute__((ext_vector_type(16)));
typedef __bf16 v8bf  __attribute__((ext_vector_type(8)));
typedef float  v8f   __attribute__((ext_vector_type(8)));

__device__ __forceinline__ __bf16 f2bf(float f) {
    unsigned u = __builtin_bit_cast(unsigned, f);
    unsigned r = u + 0x7FFFu + ((u >> 16) & 1u);   // round-to-nearest-even
    unsigned short h = (unsigned short)(r >> 16);
    return __builtin_bit_cast(__bf16, h);
}

__device__ __forceinline__ v8f wmma_bf16(v16bf a, v16bf b, v8f c) {
    return __builtin_amdgcn_wmma_f32_16x16x32_bf16(
        false, a, false, b, (short)0, c, false, false);
}

// Async DMA: 16 bytes global -> LDS per lane, tracked by ASYNCcnt.
__device__ __forceinline__ void async_b128(const __bf16* gptr, __bf16* lptr) {
    unsigned lds = (unsigned)(size_t)lptr;          // low 32 bits = LDS offset
    unsigned long long ga = (unsigned long long)(size_t)gptr;
    asm volatile("global_load_async_to_lds_b128 %0, %1, off"
                 :: "v"(lds), "v"(ga) : "memory");
}
__device__ __forceinline__ void wait_async0() {
    asm volatile("s_wait_asynccnt 0x0" ::: "memory");
}

// ---------------------------------------------------------------------------
// Weight prep: fp32 -> bf16, transposed so B-fragments are contiguous in K.
// ---------------------------------------------------------------------------
__global__ void transpose_to_bf16(const float* __restrict__ src,
                                  __bf16* __restrict__ dst, int R, int C) {
    int idx = blockIdx.x * 256 + threadIdx.x;
    if (idx >= R * C) return;
    int c = idx / R, r = idx % R;
    dst[idx] = f2bf(src[(size_t)r * C + c]);
}

__global__ void qkv_to_bf16T(const float* __restrict__ src,
                             __bf16* __restrict__ dst) {
    int idx = blockIdx.x * 256 + threadIdx.x;   // over DMODEL*DMODEL
    if (idx >= DMODEL * DMODEL) return;
    int c = idx / DMODEL, d = idx % DMODEL;
    int h = c >> 6, e = c & 63;
    dst[idx] = f2bf(src[((size_t)h * DMODEL + d) * HDIM + e]);
}

__global__ void zero_bf16(__bf16* __restrict__ p, int n) {
    int i = blockIdx.x * 256 + threadIdx.x;
    if (i < n) p[i] = f2bf(0.0f);
}

// ---------------------------------------------------------------------------
// LayerNorm (biased std, matching reference): out = g*(x-m)/(sd+1e-9)+o, bf16
// ---------------------------------------------------------------------------
__global__ __launch_bounds__(256) void layernorm_bf16(
        const float* __restrict__ X, const float* __restrict__ g,
        const float* __restrict__ ofs, __bf16* __restrict__ out) {
    const int row = blockIdx.x;
    const int tid = threadIdx.x;
    const float* x = X + (size_t)row * DMODEL;
    float s1 = 0.f, s2 = 0.f;
    for (int c = tid; c < DMODEL; c += 256) { float v = x[c]; s1 += v; s2 += v * v; }
    __shared__ float r1[256], r2[256];
    r1[tid] = s1; r2[tid] = s2; __syncthreads();
    for (int st = 128; st > 0; st >>= 1) {
        if (tid < st) { r1[tid] += r1[tid + st]; r2[tid] += r2[tid + st]; }
        __syncthreads();
    }
    float mean = r1[0] * (1.f / DMODEL);
    float var  = r2[0] * (1.f / DMODEL) - mean * mean;
    float inv  = 1.f / (sqrtf(fmaxf(var, 0.f)) + 1e-9f);
    __bf16* orow = out + (size_t)row * DMODEL;
    for (int c = tid; c < DMODEL; c += 256)
        orow[c] = f2bf(g[c] * (x[c] - mean) * inv + ofs[c]);
}

// ---------------------------------------------------------------------------
// WMMA GEMM, LDS double-buffered via async global->LDS DMA.
// C[M,N] = A[MxK](bf16) * B[KxN] given BT[NxK] bf16.
// Workgroup: 256 thr = 8 waves -> 128x64 tile; wave w owns rows w*16..w*16+15.
// Inner loop: gather 10 ds_load_b128 fragments, single dscnt wait, 4x WMMA.
// Epilogue modes: 0 f32 (+bias,+residual), 2 bf16 relu(+bias),
//                 3 bf16 -> Q/K [B,H,S,64], 4 bf16 -> Vt [B,H,64,SPAD]
// ---------------------------------------------------------------------------
__global__ __launch_bounds__(256) void gemm_bf16_wmma(
        const __bf16* __restrict__ A, const __bf16* __restrict__ BT,
        const float* __restrict__ bias, const float* __restrict__ residual,
        void* __restrict__ outp, int M, int N, int K, int mode) {
    const int tid    = threadIdx.x;
    const int lane   = tid & 31;
    const int wave   = tid >> 5;
    const int laneHi = lane >> 4;
    const int ln     = lane & 15;
    const int mt     = blockIdx.y * TILE_M;          // tile row base
    const int n0     = blockIdx.x * TILE_N;          // tile col base

    __shared__ __align__(16) __bf16 As[2][TILE_M * LDS_PITCH];
    __shared__ __align__(16) __bf16 Bs[2][TILE_N * LDS_PITCH];

    // Stage one TILE_K slab of A (128x64) and BT (64x64) into LDS buffer `buf`.
    auto issue_tile = [&](int buf, int k0) {
        // A: 128 rows x 8 b128-chunks = 1024 chunks, 4 per thread
#pragma unroll
        for (int i = 0; i < 4; ++i) {
            int idx = tid + 256 * i;
            int r = idx >> 3, q = idx & 7;
            async_b128(A + (size_t)(mt + r) * K + k0 + q * 8,
                       &As[buf][r * LDS_PITCH + q * 8]);
        }
        // B: 64 rows x 8 chunks = 512 chunks, 2 per thread
#pragma unroll
        for (int i = 0; i < 2; ++i) {
            int idx = tid + 256 * i;
            int r = idx >> 3, q = idx & 7;
            async_b128(BT + (size_t)(n0 + r) * K + k0 + q * 8,
                       &Bs[buf][r * LDS_PITCH + q * 8]);
        }
    };

    v8f acc[4];
#pragma unroll
    for (int j = 0; j < 4; j++)
#pragma unroll
        for (int i = 0; i < 8; i++) acc[j][i] = 0.f;

    const int nst = K / TILE_K;
    issue_tile(0, 0);
    wait_async0();
    __syncthreads();

    for (int st = 0; st < nst; ++st) {
        const int buf = st & 1;
        if (st + 1 < nst) issue_tile(buf ^ 1, (st + 1) * TILE_K);

        const __bf16* Arow = &As[buf][(wave * 16 + ln) * LDS_PITCH];
        const __bf16* Bbase = &Bs[buf][ln * LDS_PITCH + laneHi * 16];
#pragma unroll
        for (int kk = 0; kk < TILE_K; kk += 32) {
            // Gather ALL fragments for this K-step first (10x ds_load_b128),
            // so only one dscnt wait gates the 4-WMMA chain.
            v8bf alo = *(const v8bf*)(Arow + kk + laneHi * 16);
            v8bf ahi = *(const v8bf*)(Arow + kk + laneHi * 16 + 8);
            v16bf bfr[4];
#pragma unroll
            for (int j = 0; j < 4; j++)
                bfr[j] = *(const v16bf*)(Bbase + j * 16 * LDS_PITCH + kk);
            v16bf af;
#pragma unroll
            for (int i = 0; i < 8; i++) { af[i] = alo[i]; af[i + 8] = ahi[i]; }
#pragma unroll
            for (int j = 0; j < 4; j++)
                acc[j] = wmma_bf16(af, bfr[j], acc[j]);
        }
        wait_async0();       // next-stage DMA landed (per-wave)
        __syncthreads();     // all waves done reading `buf` + all DMA visible
    }

    const int m0 = mt + wave * 16;
#pragma unroll
    for (int j = 0; j < 4; j++) {
        int n = n0 + j * 16 + ln;
        float bv = bias ? bias[n] : 0.f;
#pragma unroll
        for (int i = 0; i < 8; i++) {
            int m = m0 + i + 8 * laneHi;       // D layout: row = i + 8*laneHi
            float v = acc[j][i] + bv;
            if (mode == 0) {
                size_t idx = (size_t)m * N + n;
                if (residual) v += residual[idx];
                ((float*)outp)[idx] = v;
            } else if (mode == 2) {
                ((__bf16*)outp)[(size_t)m * N + n] = f2bf(v > 0.f ? v : 0.f);
            } else if (mode == 3) {
                int bb = m >> 11, s = m & (SEQ - 1), hh = n >> 6, e = n & 63;
                ((__bf16*)outp)[(((size_t)bb * NHEAD + hh) * SEQ + s) * HDIM + e] = f2bf(v);
            } else { // mode 4: Vt
                int bb = m >> 11, s = m & (SEQ - 1), hh = n >> 6, e = n & 63;
                ((__bf16*)outp)[(((size_t)bb * NHEAD + hh) * HDIM + e) * SPAD + s] = f2bf(v);
            }
        }
    }
}

// ---------------------------------------------------------------------------
// Flash attention: one wave per (b, h, 16 query rows). Online softmax,
// scores & P·V via WMMA; P reshaped D-layout -> A-layout through LDS.
// ---------------------------------------------------------------------------
__global__ __launch_bounds__(32) void attn_flash(
        const __bf16* __restrict__ Qh, const __bf16* __restrict__ Kh,
        const __bf16* __restrict__ Vt, __bf16* __restrict__ ctx) {
    const int lane   = threadIdx.x;
    const int laneHi = lane >> 4;
    const int ln     = lane & 15;
    const int s0     = blockIdx.x * 16;
    const int h      = blockIdx.y;
    const int b      = blockIdx.z;
    const int bh     = b * NHEAD + h;

    __shared__ __align__(32) float  sc[16 * 32];
    __shared__ __align__(32) __bf16 pb[16 * 32];
    __shared__ float arow[16];
    __shared__ float lrow[16];

    v16bf qf[2];
    {
        const __bf16* qb = Qh + ((size_t)bh * SEQ + s0 + ln) * HDIM;
#pragma unroll
        for (int eh = 0; eh < 2; eh++) {
            v8bf lo = *(const v8bf*)(qb + eh * 32 + laneHi * 16);
            v8bf hi = *(const v8bf*)(qb + eh * 32 + laneHi * 16 + 8);
#pragma unroll
            for (int i = 0; i < 8; i++) { qf[eh][i] = lo[i]; qf[eh][i + 8] = hi[i]; }
        }
    }

    v8f o[4];
#pragma unroll
    for (int j = 0; j < 4; j++)
#pragma unroll
        for (int i = 0; i < 8; i++) o[j][i] = 0.f;

    float mloc = -3.0e38f, lloc = 0.f;
    const float scale = 0.125f;                     // 1/sqrt(64)

    for (int t0 = 0; t0 < s0 + 16; t0 += 32) {      // causal upper bound
#pragma unroll
        for (int f = 0; f < 2; f++) {
            int t  = t0 + f * 16 + ln;
            int tc = t < SEQ ? t : SEQ - 1;         // clamp reads; mask below
            // both K fragments first, then the 2-WMMA chain
            v16bf kf0 = *(const v16bf*)(Kh + ((size_t)bh * SEQ + tc) * HDIM
                                        + laneHi * 16);
            v16bf kf1 = *(const v16bf*)(Kh + ((size_t)bh * SEQ + tc) * HDIM
                                        + 32 + laneHi * 16);
            v8f c;
#pragma unroll
            for (int i = 0; i < 8; i++) c[i] = 0.f;
            c = wmma_bf16(qf[0], kf0, c);
            c = wmma_bf16(qf[1], kf1, c);
#pragma unroll
            for (int i = 0; i < 8; i++) {
                int srow = s0 + i + 8 * laneHi;
                float v = c[i] * scale;
                if (t > srow) v = -1.0e9f;          // causal masked_fill
                sc[(i + 8 * laneHi) * 32 + f * 16 + ln] = v;
            }
        }
        __syncthreads();
        if (lane < 16) {
            float tmax = -3.0e38f;
            for (int j2 = 0; j2 < 32; j2++) tmax = fmaxf(tmax, sc[lane * 32 + j2]);
            float mnew  = fmaxf(mloc, tmax);
            float alpha = __expf(mloc - mnew);
            float ssum  = 0.f;
            for (int j2 = 0; j2 < 32; j2++) {
                float p = __expf(sc[lane * 32 + j2] - mnew);
                pb[lane * 32 + j2] = f2bf(p);
                ssum += p;
            }
            lloc = lloc * alpha + ssum;
            mloc = mnew;
            arow[lane] = alpha;
        }
        __syncthreads();
#pragma unroll
        for (int i = 0; i < 8; i++) {
            float a = arow[i + 8 * laneHi];
            o[0][i] *= a; o[1][i] *= a; o[2][i] *= a; o[3][i] *= a;
        }
        v16bf pf;
        {
            v8bf lo = *(const v8bf*)(pb + ln * 32 + laneHi * 16);
            v8bf hi = *(const v8bf*)(pb + ln * 32 + laneHi * 16 + 8);
#pragma unroll
            for (int i = 0; i < 8; i++) { pf[i] = lo[i]; pf[i + 8] = hi[i]; }
        }
        v16bf vfr[4];
#pragma unroll
        for (int j = 0; j < 4; j++)
            vfr[j] = *(const v16bf*)(Vt + ((size_t)bh * HDIM + j * 16 + ln) * SPAD
                                     + t0 + laneHi * 16);
#pragma unroll
        for (int j = 0; j < 4; j++)
            o[j] = wmma_bf16(pf, vfr[j], o[j]);
        __syncthreads();
    }

    if (lane < 16) lrow[lane] = lloc;
    __syncthreads();
#pragma unroll
    for (int j = 0; j < 4; j++)
#pragma unroll
        for (int i = 0; i < 8; i++) {
            int srow = s0 + i + 8 * laneHi;
            float v = o[j][i] / lrow[i + 8 * laneHi];
            size_t r = (size_t)b * SEQ + srow;
            ctx[r * DMODEL + h * HDIM + j * 16 + ln] = f2bf(v);
        }
}

// ---------------------------------------------------------------------------
// Host-side launcher
// ---------------------------------------------------------------------------
extern "C" void kernel_launch(void* const* d_in, const int* in_sizes, int n_in,
                              void* d_out, int out_size, void* d_ws, size_t ws_size,
                              hipStream_t stream) {
    const float* X  = (const float*)d_in[0];
    const float* Wq = (const float*)d_in[2];
    const float* bq = (const float*)d_in[3];
    const float* Wk = (const float*)d_in[4];
    const float* bk = (const float*)d_in[5];
    const float* Wv = (const float*)d_in[6];
    const float* bv = (const float*)d_in[7];
    const float* W0 = (const float*)d_in[8];
    const float* b0 = (const float*)d_in[9];
    const float* W1 = (const float*)d_in[10];
    const float* b1 = (const float*)d_in[11];
    const float* W2 = (const float*)d_in[12];
    const float* b2 = (const float*)d_in[13];
    const float* g1 = (const float*)d_in[14];
    const float* o1 = (const float*)d_in[15];
    const float* g2 = (const float*)d_in[16];
    const float* o2 = (const float*)d_in[17];
    float* out = (float*)d_out;

    char* w = (char*)d_ws;
    size_t off = 0;
    auto alloc = [&](size_t bytes) -> void* {
        void* p = w + off;
        off += (bytes + 255) & ~(size_t)255;
        return p;
    };
    __bf16* Xn  = (__bf16*)alloc((size_t)MROWS * DMODEL * 2);
    __bf16* WqT = (__bf16*)alloc((size_t)DMODEL * DMODEL * 2);
    __bf16* WkT = (__bf16*)alloc((size_t)DMODEL * DMODEL * 2);
    __bf16* WvT = (__bf16*)alloc((size_t)DMODEL * DMODEL * 2);
    __bf16* W0T = (__bf16*)alloc((size_t)DMODEL * DMODEL * 2);
    __bf16* W1T = (__bf16*)alloc((size_t)DMODEL * DFFN * 2);
    __bf16* W2T = (__bf16*)alloc((size_t)DFFN * DMODEL * 2);
    __bf16* Q_  = (__bf16*)alloc((size_t)BATCH * NHEAD * SEQ * HDIM * 2);
    __bf16* K_  = (__bf16*)alloc((size_t)BATCH * NHEAD * SEQ * HDIM * 2);
    __bf16* Vt  = (__bf16*)alloc((size_t)BATCH * NHEAD * HDIM * SPAD * 2);
    __bf16* Ctx = (__bf16*)alloc((size_t)MROWS * DMODEL * 2);
    float*  Y   = (float*)alloc((size_t)MROWS * DMODEL * 4);
    __bf16* H2n = (__bf16*)alloc((size_t)MROWS * DMODEL * 2);
    __bf16* FF  = (__bf16*)alloc((size_t)MROWS * DFFN * 2);
    (void)ws_size; (void)in_sizes; (void)n_in; (void)out_size;

    const int DD = DMODEL * DMODEL;        // 1M
    const int DW = DMODEL * DFFN;          // 4M
    qkv_to_bf16T<<<(DD + 255) / 256, 256, 0, stream>>>(Wq, WqT);
    qkv_to_bf16T<<<(DD + 255) / 256, 256, 0, stream>>>(Wk, WkT);
    qkv_to_bf16T<<<(DD + 255) / 256, 256, 0, stream>>>(Wv, WvT);
    transpose_to_bf16<<<(DD + 255) / 256, 256, 0, stream>>>(W0, W0T, DMODEL, DMODEL);
    transpose_to_bf16<<<(DW + 255) / 256, 256, 0, stream>>>(W1, W1T, DMODEL, DFFN);
    transpose_to_bf16<<<(DW + 255) / 256, 256, 0, stream>>>(W2, W2T, DFFN, DMODEL);
    {
        int nvt = BATCH * NHEAD * HDIM * SPAD;
        zero_bf16<<<(nvt + 255) / 256, 256, 0, stream>>>(Vt, nvt);
    }
    layernorm_bf16<<<MROWS, 256, 0, stream>>>(X, g1, o1, Xn);

    dim3 gQ(DMODEL / TILE_N, MROWS / TILE_M);          // (16, 32)
    gemm_bf16_wmma<<<gQ, 256, 0, stream>>>(Xn, WqT, bq, nullptr, Q_,
                                           MROWS, DMODEL, DMODEL, 3);
    gemm_bf16_wmma<<<gQ, 256, 0, stream>>>(Xn, WkT, bk, nullptr, K_,
                                           MROWS, DMODEL, DMODEL, 3);
    gemm_bf16_wmma<<<gQ, 256, 0, stream>>>(Xn, WvT, bv, nullptr, Vt,
                                           MROWS, DMODEL, DMODEL, 4);

    dim3 gA(SEQ / 16, NHEAD, BATCH);
    attn_flash<<<gA, 32, 0, stream>>>(Q_, K_, Vt, Ctx);

    gemm_bf16_wmma<<<gQ, 256, 0, stream>>>(Ctx, W0T, b0, X, Y,
                                           MROWS, DMODEL, DMODEL, 0);
    layernorm_bf16<<<MROWS, 256, 0, stream>>>(Y, g2, o2, H2n);

    dim3 gF1(DFFN / TILE_N, MROWS / TILE_M);           // (64, 32)
    gemm_bf16_wmma<<<gF1, 256, 0, stream>>>(H2n, W1T, b1, nullptr, FF,
                                            MROWS, DFFN, DMODEL, 2);
    gemm_bf16_wmma<<<gQ, 256, 0, stream>>>(FF, W2T, b2, Y, out,
                                           MROWS, DMODEL, DFFN, 0);
}
